// AutoEncoderTopK_53446573031859
// MI455X (gfx1250) — matrix operand order repaired and checked by reading, not compiled, not run on hardware
//
#include <hip/hip_runtime.h>
#include <hip/hip_bf16.h>
#include <math.h>

#define ACT_DIM   768
#define DICT_SIZE 24576
#define BATCH     8192
#define TOPK      64
#define NSEG      4
#define SEG_DICT  (DICT_SIZE / NSEG)      // 6144
#define NIT       (SEG_DICT / 16)         // 384 tiles of 16 dict entries
#define KSTEPS    (ACT_DIM / 32)          // 24 WMMA k-steps

typedef __attribute__((ext_vector_type(16))) __bf16 v16bf;
typedef __attribute__((ext_vector_type(8)))  __bf16 v8bf;
typedef __attribute__((ext_vector_type(8)))  float  v8f;

// ---------------------------------------------------------------------------
// float -> bf16 round-to-nearest-even
// ---------------------------------------------------------------------------
__device__ __forceinline__ unsigned short f2bf(float f) {
  unsigned u = __float_as_uint(f);
  u += 0x7FFFu + ((u >> 16) & 1u);
  return (unsigned short)(u >> 16);
}

// ---------------------------------------------------------------------------
// Kernel 1: xb[b][k] = bf16(x[b][k] - b_dec[k])
// ---------------------------------------------------------------------------
__global__ void cvt_x_kernel(const float* __restrict__ x,
                             const float* __restrict__ b_dec,
                             unsigned short* __restrict__ out) {
  size_t i = ((size_t)blockIdx.x * blockDim.x + threadIdx.x) * 4;
  if (i >= (size_t)BATCH * ACT_DIM) return;
  int col = (int)(i % ACT_DIM);
  float4 v = *(const float4*)(x + i);
  float4 b = *(const float4*)(b_dec + col);
  ushort4 o;
  o.x = f2bf(v.x - b.x);
  o.y = f2bf(v.y - b.y);
  o.z = f2bf(v.z - b.z);
  o.w = f2bf(v.w - b.w);
  *(ushort4*)(out + i) = o;
}

// ---------------------------------------------------------------------------
// Kernel 2: wb[i] = bf16(W_enc[i])
// ---------------------------------------------------------------------------
__global__ void cvt_w_kernel(const float* __restrict__ w,
                             unsigned short* __restrict__ out) {
  size_t i = ((size_t)blockIdx.x * blockDim.x + threadIdx.x) * 4;
  if (i >= (size_t)DICT_SIZE * ACT_DIM) return;
  float4 v = *(const float4*)(w + i);
  ushort4 o;
  o.x = f2bf(v.x);
  o.y = f2bf(v.y);
  o.z = f2bf(v.z);
  o.w = f2bf(v.w);
  *(ushort4*)(out + i) = o;
}

// ---------------------------------------------------------------------------
// Kernel 3: WdT[d][a] = W_dec[a][d]
// ---------------------------------------------------------------------------
__global__ void transpose_kernel(const float* __restrict__ in,
                                 float* __restrict__ outT) {
  __shared__ float tile[32][33];
  const int bx = blockIdx.x, by = blockIdx.y;
  const int tx = threadIdx.x, ty = threadIdx.y;      // block (32,8)
  const int xd = bx * 32 + tx;
  #pragma unroll
  for (int i = ty; i < 32; i += 8)
    tile[i][tx] = in[(size_t)(by * 32 + i) * DICT_SIZE + xd];
  __syncthreads();
  const int xa = by * 32 + tx;
  #pragma unroll
  for (int i = ty; i < 32; i += 8)
    outT[(size_t)(bx * 32 + i) * ACT_DIM + xa] = tile[tx][i];
}

// ---------------------------------------------------------------------------
// Kernel 4: encoder GEMM (bf16 WMMA) + ReLU + per-segment top-64.
// Grid: (BATCH/128, NSEG), block 256 (8 waves). Each wave owns 16 rows with
// A register-resident; all 8 waves share a double-buffered 16x768 bf16 W_enc
// tile in LDS. B fragments are software-pipelined 2 deep with hard
// sched_barrier(0) fences so RA cannot collapse the rotation back to one
// register set (goal: s_wait_dscnt <= 2..4 instead of a full drain per WMMA).
// ---------------------------------------------------------------------------
__global__ __launch_bounds__(256) void sae_encode_kernel(
    const unsigned short* __restrict__ xbu,   // [BATCH][ACT_DIM] bf16
    const unsigned short* __restrict__ wbu,   // [DICT_SIZE][ACT_DIM] bf16
    const float* __restrict__ b_enc,
    float* __restrict__ topv_seg,             // [BATCH][NSEG][64]
    int*   __restrict__ topi_seg) {           // [BATCH][NSEG][64]
  __shared__ uint4 bufB[2][1536];             // 2 x 24KB W_enc tile
  __shared__ float topvS[8][16][64];          // 32KB
  __shared__ int   topiS[8][16][64];          // 32KB
  __shared__ float thrS[8][16];

  const int tid  = threadIdx.x;
  const int lane = tid & 31;
  const int wave = tid >> 5;
  const int m    = lane & 15;
  const int half = lane >> 4;
  const int segbase = blockIdx.y * SEG_DICT;
  const int rowb = blockIdx.x * 128 + wave * 16;

  // ---- A fragments (16x32 bf16 each), register resident for all 24 k-steps
  v16bf afrag[KSTEPS];
  {
    const __bf16* arow = (const __bf16*)xbu + (size_t)(rowb + m) * ACT_DIM;
    #pragma unroll
    for (int kk = 0; kk < KSTEPS; ++kk) {
      const int kb = kk * 32;
      v8bf alo = *(const v8bf*)(arow + kb + 16 * half);
      v8bf ahi = *(const v8bf*)(arow + kb + 16 * half + 8);
      afrag[kk] = __builtin_shufflevector(alo, ahi,
                    0,1,2,3,4,5,6,7,8,9,10,11,12,13,14,15);
    }
  }

  // ---- init per-wave top-k state
  #pragma unroll
  for (int i = 0; i < 32; ++i) {
    int e = lane + 32 * i;                    // 0..1023
    topvS[wave][e >> 6][e & 63] = -INFINITY;
    topiS[wave][e >> 6][e & 63] = 0;
  }
  if (lane < 16) thrS[wave][lane] = -INFINITY;

  // ---- preload W_enc tile 0
  {
    const uint4* bs = (const uint4*)(wbu + (size_t)segbase * ACT_DIM);
    #pragma unroll
    for (int j = 0; j < 6; ++j) bufB[0][tid + 256 * j] = bs[tid + 256 * j];
  }
  __syncthreads();

  for (int it = 0; it < NIT; ++it) {
    const int p = it & 1;
    const int dbase = segbase + it * 16;
    const bool have_next = (it + 1 < NIT);

    // issue next tile's global loads early (overlap with WMMAs)
    uint4 st[6];
    if (have_next) {
      const uint4* bs = (const uint4*)(wbu + (size_t)(dbase + 16) * ACT_DIM);
      #pragma unroll
      for (int j = 0; j < 6; ++j) st[j] = bs[tid + 256 * j];
    }

    // ---- 24 WMMAs, two accumulator chains, B prefetch depth 2 enforced
    //      by hard scheduling fences.
    const __bf16* brow = (const __bf16*)(&bufB[p][0]) + m * ACT_DIM;
    // B fragment (32x16 bf16): VGPR0-3 K=16*half.., VGPR4-7 K=16*half+8..
    v8bf lo_a = *(const v8bf*)(brow + 0 * 32 + 16 * half);
    v8bf hi_a = *(const v8bf*)(brow + 0 * 32 + 16 * half + 8);
    v8bf lo_b = *(const v8bf*)(brow + 1 * 32 + 16 * half);
    v8bf hi_b = *(const v8bf*)(brow + 1 * 32 + 16 * half + 8);
    __builtin_amdgcn_sched_barrier(0);
    v8f c0 = {}, c1 = {};
    #pragma unroll
    for (int kk = 0; kk < KSTEPS; ++kk) {
      v16bf b = __builtin_shufflevector(lo_a, hi_a,
                  0,1,2,3,4,5,6,7,8,9,10,11,12,13,14,15);
      v8bf nlo = lo_b, nhi = hi_b;            // dead when kk+2 >= KSTEPS
      if (kk + 2 < KSTEPS) {
        const int kb = (kk + 2) * 32;
        nlo = *(const v8bf*)(brow + kb + 16 * half);
        nhi = *(const v8bf*)(brow + kb + 16 * half + 8);
      }
      if (kk & 1)
        c1 = __builtin_amdgcn_wmma_f32_16x16x32_bf16(false, afrag[kk], false, b,
                                                     (short)0, c1, false, false);
      else
        c0 = __builtin_amdgcn_wmma_f32_16x16x32_bf16(false, afrag[kk], false, b,
                                                     (short)0, c0, false, false);
      lo_a = lo_b; hi_a = hi_b;
      lo_b = nlo;  hi_b = nhi;
      __builtin_amdgcn_sched_barrier(0);
    }

    // epilogue: + b_enc, ReLU.  c[r] = row (r + 8*half), col (dbase + m)
    const float be = b_enc[dbase + m];
    v8f c;
    #pragma unroll
    for (int r = 0; r < 8; ++r) c[r] = fmaxf(c0[r] + c1[r] + be, 0.0f);

    // ---- streaming per-wave top-64 (threshold-gated cooperative inserts)
    #pragma unroll 1
    for (int r = 0; r < 8; ++r) {
      const float cv  = c[r];
      const float thr = thrS[wave][r + 8 * half];
      unsigned long long mask = __ballot(cv > thr) & 0xffffffffull;
      while (mask) {
        const int srcl = (int)__builtin_ctzll(mask);
        mask &= mask - 1;
        const float val  = __shfl(cv, srcl, 32);
        const int   row  = r + ((srcl >> 4) << 3);
        const int   gidx = dbase + (srcl & 15);
        float v0 = topvS[wave][row][lane];
        float v1 = topvS[wave][row][lane + 32];
        float lv; int le;
        if (v0 <= v1) { lv = v0; le = lane; } else { lv = v1; le = lane + 32; }
        #pragma unroll
        for (int off = 16; off > 0; off >>= 1) {          // argmin butterfly
          float ov = __shfl_xor(lv, off, 32);
          int   oe = __shfl_xor(le, off, 32);
          if (ov < lv || (ov == lv && oe < le)) { lv = ov; le = oe; }
        }
        if (val > lv) {
          if (lane == (le & 31)) {
            topvS[wave][row][le] = val;
            topiS[wave][row][le] = gidx;
            if (le < 32) v0 = val; else v1 = val;
          }
          float nm = fminf(v0, v1);
          #pragma unroll
          for (int off = 16; off > 0; off >>= 1)
            nm = fminf(nm, __shfl_xor(nm, off, 32));
          if (lane == 0) thrS[wave][row] = nm;
        }
      }
    }

    // ---- commit staged tile, flip buffers
    if (have_next) {
      #pragma unroll
      for (int j = 0; j < 6; ++j) bufB[p ^ 1][tid + 256 * j] = st[j];
    }
    __syncthreads();
  }

  // ---- write segment top-64 lists
  #pragma unroll
  for (int rr = 0; rr < 16; ++rr) {
    const size_t base = ((size_t)(rowb + rr) * NSEG + blockIdx.y) * 64;
    #pragma unroll
    for (int e = lane; e < 64; e += 32) {
      topv_seg[base + e] = topvS[wave][rr][e];
      topi_seg[base + e] = topiS[wave][rr][e];
    }
  }
}

// ---------------------------------------------------------------------------
// Kernel 5: merge 4 segment top-64 lists -> global top-64, then sparse decode.
// One wave per row.
// ---------------------------------------------------------------------------
__global__ __launch_bounds__(32) void sae_merge_decode_kernel(
    const float* __restrict__ topv_seg,       // [BATCH][NSEG*64]
    const int*   __restrict__ topi_seg,
    const float* __restrict__ WdT,            // [DICT_SIZE][ACT_DIM]
    const float* __restrict__ b_dec,
    float* __restrict__ out) {
  __shared__ float selv[64];
  __shared__ int   seli[64];
  const int lane = threadIdx.x;
  const int row  = blockIdx.x;

  // 256 candidates, 8 per lane
  float bv[8]; int bi[8];
  {
    const float* tv = topv_seg + (size_t)row * (NSEG * 64);
    const int*   ti = topi_seg + (size_t)row * (NSEG * 64);
    #pragma unroll
    for (int j = 0; j < 8; ++j) {
      bv[j] = tv[lane + 32 * j];
      bi[j] = ti[lane + 32 * j];
    }
  }
  selv[lane] = 0.0f; selv[lane + 32] = 0.0f;
  seli[lane] = 0;    seli[lane + 32] = 0;

  // 64 rounds of wave-cooperative argmax-extract
  for (int s = 0; s < 64; ++s) {
    float lv = bv[0]; int li = bi[0]; int lslot = 0;
    #pragma unroll
    for (int j = 1; j < 8; ++j)
      if (bv[j] > lv) { lv = bv[j]; li = bi[j]; lslot = j; }
    int lown = (lane << 3) | lslot;
    #pragma unroll
    for (int off = 16; off > 0; off >>= 1) {
      float ov = __shfl_xor(lv, off, 32);
      int   oi = __shfl_xor(li, off, 32);
      int   oo = __shfl_xor(lown, off, 32);
      if (ov > lv || (ov == lv && oo < lown)) { lv = ov; li = oi; lown = oo; }
    }
    if (!(lv > 0.0f)) break;                  // remaining contribute nothing
    if (lane == 0) { selv[s] = lv; seli[s] = li; }
    if ((lown >> 3) == lane) {
      const int wslot = lown & 7;
      #pragma unroll
      for (int j = 0; j < 8; ++j)
        if (j == wslot) bv[j] = -INFINITY;
    }
  }

  // sparse decode: x_hat = sum val_k * WdT[idx_k][:] + b_dec
  float4 acc[6];
  const float4* bd = (const float4*)b_dec;
  #pragma unroll
  for (int j = 0; j < 6; ++j) acc[j] = bd[lane + 32 * j];
  for (int k = 0; k < 64; ++k) {
    const float val = selv[k];
    if (!(val > 0.0f)) continue;
    const int col = seli[k];
    const float4* w = (const float4*)(WdT + (size_t)col * ACT_DIM);
    #pragma unroll
    for (int j = 0; j < 6; ++j) {
      float4 t = w[lane + 32 * j];
      acc[j].x = fmaf(val, t.x, acc[j].x);
      acc[j].y = fmaf(val, t.y, acc[j].y);
      acc[j].z = fmaf(val, t.z, acc[j].z);
      acc[j].w = fmaf(val, t.w, acc[j].w);
    }
  }
  float4* o = (float4*)(out + (size_t)row * ACT_DIM);
  #pragma unroll
  for (int j = 0; j < 6; ++j) o[lane + 32 * j] = acc[j];
}

// ---------------------------------------------------------------------------
extern "C" void kernel_launch(void* const* d_in, const int* in_sizes, int n_in,
                              void* d_out, int out_size, void* d_ws, size_t ws_size,
                              hipStream_t stream) {
  (void)in_sizes; (void)n_in; (void)out_size; (void)ws_size;
  const float* x     = (const float*)d_in[0];
  const float* W_enc = (const float*)d_in[1];
  const float* b_enc = (const float*)d_in[2];
  const float* W_dec = (const float*)d_in[3];
  const float* b_dec = (const float*)d_in[4];
  float* out = (float*)d_out;

  char* ws = (char*)d_ws;
  const size_t xb_bytes  = (size_t)BATCH * ACT_DIM * sizeof(unsigned short);
  const size_t wb_bytes  = (size_t)DICT_SIZE * ACT_DIM * sizeof(unsigned short);
  const size_t wdt_bytes = (size_t)DICT_SIZE * ACT_DIM * sizeof(float);
  const size_t tv_bytes  = (size_t)BATCH * NSEG * 64 * sizeof(float);
  unsigned short* xb  = (unsigned short*)ws;
  unsigned short* wb  = (unsigned short*)(ws + xb_bytes);
  float*          WdT = (float*)(ws + xb_bytes + wb_bytes);
  float*          tvs = (float*)(ws + xb_bytes + wb_bytes + wdt_bytes);
  int*            tis = (int*)(ws + xb_bytes + wb_bytes + wdt_bytes + tv_bytes);

  {
    size_t n = (size_t)BATCH * ACT_DIM / 4;
    cvt_x_kernel<<<(unsigned)((n + 255) / 256), 256, 0, stream>>>(x, b_dec, xb);
  }
  {
    size_t n = (size_t)DICT_SIZE * ACT_DIM / 4;
    cvt_w_kernel<<<(unsigned)((n + 255) / 256), 256, 0, stream>>>(W_enc, wb);
  }
  {
    dim3 g(DICT_SIZE / 32, ACT_DIM / 32);
    dim3 b(32, 8);
    transpose_kernel<<<g, b, 0, stream>>>(W_dec, WdT);
  }
  {
    dim3 g(BATCH / 128, NSEG);
    sae_encode_kernel<<<g, 256, 0, stream>>>(xb, wb, b_enc, tvs, tis);
  }
  sae_merge_decode_kernel<<<BATCH, 32, 0, stream>>>(tvs, tis, WdT, b_dec, out);
}
